// CustomRNN_1511828488500
// MI455X (gfx1250) — matrix operand order repaired and checked by reading, not compiled, not run on hardware
//
#include <hip/hip_runtime.h>
#include <hip/hip_bf16.h>

#define BB 128
#define TT 256
#define DD 256
#define HH 256

typedef __attribute__((ext_vector_type(16))) __bf16 v16bf;
typedef __attribute__((ext_vector_type(8)))  __bf16 v8bf;
typedef __attribute__((ext_vector_type(8)))  float  v8f;

// ---------------- WMMA fragment helpers (wave32, 16x16x32 bf16) ----------------

// B matrix 32x16 (KxN), fp32 source, row-major, leading dim ldw (slow path for
// the fully parallel kernels; the serial scan uses the packed path below).
// Layout: lanes 0-15 hold K=0..15 (N = n0 + (lane&15)), lanes 16-31 hold K=16..31;
// element j of the lane vector holds K = (lane&16) + j.
__device__ __forceinline__ v16bf load_b_frag(const float* __restrict__ W, int ldw,
                                             int k0, int n0, int lane) {
  v16bf b;
  const int n  = n0 + (lane & 15);
  const int kb = k0 + (lane & 16);
#pragma unroll
  for (int v = 0; v < 8; ++v) {
    b[2 * v + 0] = (__bf16)W[(size_t)(kb + 2 * v + 0) * ldw + n];
    b[2 * v + 1] = (__bf16)W[(size_t)(kb + 2 * v + 1) * ldw + n];
  }
  return b;
}

// Same, but B is the transpose of W: element (k,n) = W[n*ldw + k]  (for fc_w.T)
__device__ __forceinline__ v16bf load_b_frag_T(const float* __restrict__ W, int ldw,
                                               int k0, int n0, int lane) {
  v16bf b;
  const int n  = n0 + (lane & 15);
  const int kb = k0 + (lane & 16);
#pragma unroll
  for (int v = 0; v < 8; ++v) {
    b[2 * v + 0] = (__bf16)W[(size_t)n * ldw + kb + 2 * v + 0];
    b[2 * v + 1] = (__bf16)W[(size_t)n * ldw + kb + 2 * v + 1];
  }
  return b;
}

// Packed-B fast path: a 32x16 tile is stored as 1KB in exact fragment order:
// chunk0 (256 bf16) = per-lane 16B holding K=(lane&16)+0..7, chunk1 = K=+8..15.
// One fragment = two perfectly coalesced 512B wave loads (global_load_b128).
__device__ __forceinline__ v16bf load_b_packed(const __bf16* __restrict__ base, int lane) {
  const v8bf c0 = *(const v8bf*)(base + lane * 8);
  const v8bf c1 = *(const v8bf*)(base + 256 + lane * 8);
  return __builtin_shufflevector(c0, c1, 0, 1, 2, 3, 4, 5, 6, 7,
                                 8, 9, 10, 11, 12, 13, 14, 15);
}

// A matrix 16x32 (MxK), bf16 in LDS, row-major [16][aw]; lowers to 2x ds_load_b128.
__device__ __forceinline__ v16bf load_a_frag(const __bf16* A, int aw, int k0, int lane) {
  v16bf a;
  const int m  = lane & 15;
  const int kb = k0 + (lane & 16);
#pragma unroll
  for (int v = 0; v < 8; ++v) {
    const int k  = kb + 2 * v;
    a[2 * v + 0] = A[m * aw + k];
    a[2 * v + 1] = A[m * aw + k + 1];
  }
  return a;
}

#define WMMA_BF16(a, b, c) \
  __builtin_amdgcn_wmma_f32_16x16x32_bf16(false, (a), false, (b), (short)0, (c), false, false)

// ---------------- Kernel 0: pack scan weights into bf16 fragment order ----------------
// grid (T, 3): mat 0 = Whh_f[0,t], 1 = Wih_f[1,t], 2 = Whh_f[1,t].
// Coalesced fp32 row reads -> LDS scatter in packed order -> coalesced b128 writes.

__global__ __launch_bounds__(256) void pack_kernel(const float* __restrict__ Whh,
                                                   const float* __restrict__ Wih,
                                                   __bf16* __restrict__ wpk) {
  __shared__ alignas(16) __bf16 Ls[8192];  // one 32x256 k-slice, packed (16KB)
  const int t = blockIdx.x, mat = blockIdx.y;
  const int tid = threadIdx.x;
  const float* W;
  if (mat == 0)      W = Whh + (size_t)t * HH * HH;
  else if (mat == 1) W = Wih + ((size_t)TT + t) * (size_t)DD * HH;
  else               W = Whh + ((size_t)TT + t) * (size_t)HH * HH;
  __bf16* dst = wpk + ((size_t)mat * TT + t) * (size_t)(8 * 16 * 512);

  for (int ks = 0; ks < 8; ++ks) {
#pragma unroll
    for (int i = tid; i < 8192; i += 256) {
      const int k = i >> 8, n = i & 255;              // coalesced along n
      const float e = W[(size_t)(ks * 32 + k) * HH + n];
      const int nt  = n >> 4;
      const int ln  = (k & 16) + (n & 15);
      const int pos = nt * 512 + ((k >> 3) & 1) * 256 + ln * 8 + (k & 7);
      Ls[pos] = (__bf16)e;
    }
    __syncthreads();
#pragma unroll
    for (int i = tid * 8; i < 8192; i += 256 * 8) {   // coalesced 16B copies
      *(uint4*)(dst + ks * 8192 + i) = *(const uint4*)(Ls + i);
    }
    __syncthreads();
  }
}

// ---------------- Kernel 1: pre = x[:,t,:] @ Wih_f[0,t] + b_f[0,t] ----------------
// Fully parallel; grid (T, B/16), 8 waves, wave w owns output cols [32w, 32w+32).
// Output is stored in C-fragment-packed order so the scan's epilogue read is
// perfectly coalesced: pre_pk[((t*8 + bt)*16 + nt)*256 + lane*8 + r].

__global__ __launch_bounds__(256) void pre_kernel(const float* __restrict__ x,
                                                  const float* __restrict__ Wih0,
                                                  const float* __restrict__ b0,
                                                  float* __restrict__ pre) {
  __shared__ alignas(16) __bf16 As[16 * 256];
  const int t = blockIdx.x, bt = blockIdx.y, rb = bt * 16;
  const int tid = threadIdx.x, lane = tid & 31, w = tid >> 5;

  for (int i = tid; i < 16 * 256; i += 256) {
    const int r = i >> 8, c = i & 255;
    As[i] = (__bf16)x[((size_t)(rb + r) * TT + t) * DD + c];
  }
  __syncthreads();

  const float* W    = Wih0 + (size_t)t * DD * HH;
  const float* bias = b0 + (size_t)t * HH;
  const int n0 = w * 32;
  v8f acc0 = {}, acc1 = {};
#pragma unroll
  for (int k0 = 0; k0 < 256; k0 += 32) {
    v16bf a  = load_a_frag(As, 256, k0, lane);
    v16bf bA = load_b_frag(W, HH, k0, n0, lane);
    v16bf bB = load_b_frag(W, HH, k0, n0 + 16, lane);
    acc0 = WMMA_BF16(a, bA, acc0);
    acc1 = WMMA_BF16(a, bB, acc1);
  }
  const int nn = lane & 15;
  float* d0 = pre + ((((size_t)t * 8 + bt) * 16) + 2 * w) * 256 + lane * 8;
#pragma unroll
  for (int r = 0; r < 8; ++r) {
    d0[r]       = acc0[r] + bias[n0 + nn];        // tile nt = 2w
    d0[256 + r] = acc1[r] + bias[n0 + 16 + nn];   // tile nt = 2w+1
  }
}

// ---------------- Kernel 2: persistent forward scan ----------------
// B/16 blocks own disjoint batch rows -> no inter-WG sync on the serial chain.
// h0/h1 in LDS (bf16, double buffered); weights read via packed-b128 fast path.

__global__ __launch_bounds__(256) void scan_kernel(const float* __restrict__ pre,
                                                   const __bf16* __restrict__ wpk,
                                                   const float* __restrict__ bvec,
                                                   float* __restrict__ hf) {
  __shared__ alignas(16) __bf16 h0s[2][16 * 256];
  __shared__ alignas(16) __bf16 h1s[2][16 * 256];
  const int rb = blockIdx.x * 16;
  const int tid = threadIdx.x, lane = tid & 31, w = tid >> 5;
  const int n0 = w * 32;
  const int m0 = (lane >> 4) << 3, nn = lane & 15;

  for (int i = tid; i < 16 * 256; i += 256) {
    h0s[0][i] = (__bf16)0.0f; h0s[1][i] = (__bf16)0.0f;
    h1s[0][i] = (__bf16)0.0f; h1s[1][i] = (__bf16)0.0f;
  }
  __syncthreads();

  for (int t = 0; t < TT; ++t) {
    const __bf16* h0c = h0s[t & 1];
    __bf16*       h0n = h0s[(t & 1) ^ 1];
    const __bf16* h1c = h1s[t & 1];
    __bf16*       h1n = h1s[(t & 1) ^ 1];

    const __bf16* P0 = wpk + ((size_t)0 * TT + t) * 65536;  // layer0 Whh[t]
    const __bf16* P1 = wpk + ((size_t)1 * TT + t) * 65536;  // layer1 Wih[t]
    const __bf16* P2 = wpk + ((size_t)2 * TT + t) * 65536;  // layer1 Whh[t]
    const float*  b1 = bvec + ((size_t)TT + t) * HH;
    // this block's C-fragment-packed pre chunk for step t (tiles 2w, 2w+1)
    const float*  pt = pre + ((((size_t)t * 8 + blockIdx.x) * 16) + 2 * w) * 256 + lane * 8;

    // Prefetch next step's packed weights (3 x 128KB, full 256B-stride coverage).
    if (t + 1 < TT) {
      const __bf16* q0 = P0 + 65536;
      const __bf16* q1 = P1 + 65536;
      const __bf16* q2 = P2 + 65536;
#pragma unroll
      for (int i = 0; i < 2; ++i) {
        const size_t off = (size_t)(tid + i * 256) * 128;
        __builtin_prefetch(q0 + off, 0, 0);
        __builtin_prefetch(q1 + off, 0, 0);
        __builtin_prefetch(q2 + off, 0, 0);
      }
    }

    // ---- phase A: h0_new = tanh(pre[t] + h0 @ Whh0) ----
    {
      v8f acc0 = {}, acc1 = {};
#pragma unroll
      for (int ks = 0; ks < 8; ++ks) {
        v16bf a = load_a_frag(h0c, 256, ks * 32, lane);
        const __bf16* pb = P0 + (ks * 16 + (w << 1)) * 512;
        acc0 = WMMA_BF16(a, load_b_packed(pb, lane), acc0);
        acc1 = WMMA_BF16(a, load_b_packed(pb + 512, lane), acc1);
      }
#pragma unroll
      for (int r = 0; r < 8; ++r) {
        const int m = m0 + r;
        const float v0 = tanhf(acc0[r] + pt[r]);        // coalesced b128 reads
        const float v1 = tanhf(acc1[r] + pt[256 + r]);
        h0n[m * 256 + n0 + nn]      = (__bf16)v0;
        h0n[m * 256 + n0 + 16 + nn] = (__bf16)v1;
      }
    }
    __syncthreads();

    // ---- phase B: h1_new = tanh(h0_new @ Wih1 + h1 @ Whh1 + b1) ----
    {
      v8f acc0 = {}, acc1 = {};
#pragma unroll
      for (int ks = 0; ks < 8; ++ks) {
        v16bf ai = load_a_frag(h0n, 256, ks * 32, lane);
        v16bf ah = load_a_frag(h1c, 256, ks * 32, lane);
        const __bf16* p1 = P1 + (ks * 16 + (w << 1)) * 512;
        const __bf16* p2 = P2 + (ks * 16 + (w << 1)) * 512;
        acc0 = WMMA_BF16(ai, load_b_packed(p1, lane), acc0);
        acc0 = WMMA_BF16(ah, load_b_packed(p2, lane), acc0);
        acc1 = WMMA_BF16(ai, load_b_packed(p1 + 512, lane), acc1);
        acc1 = WMMA_BF16(ah, load_b_packed(p2 + 512, lane), acc1);
      }
#pragma unroll
      for (int r = 0; r < 8; ++r) {
        const int m = m0 + r;
        const float v0 = tanhf(acc0[r] + b1[n0 + nn]);
        const float v1 = tanhf(acc1[r] + b1[n0 + 16 + nn]);
        h1n[m * 256 + n0 + nn]      = (__bf16)v0;
        h1n[m * 256 + n0 + 16 + nn] = (__bf16)v1;
        if (t == TT - 1) {
          hf[(size_t)(rb + m) * HH + n0 + nn]      = v0;
          hf[(size_t)(rb + m) * HH + n0 + 16 + nn] = v1;
        }
      }
    }
    __syncthreads();
  }
}

// ---------------- Kernel 3: out = tanh(A @ W + bias), A rows strided ----------------
// Backward direction's single contributing step (h0 = 0 -> Whh term vanishes).

__global__ __launch_bounds__(256) void gemm_tanh_kernel(const float* __restrict__ A,
                                                        long strideA,
                                                        const float* __restrict__ W,
                                                        const float* __restrict__ bias,
                                                        float* __restrict__ out) {
  __shared__ alignas(16) __bf16 As[16 * 256];
  const int rb = blockIdx.x * 16;
  const int tid = threadIdx.x, lane = tid & 31, w = tid >> 5;

  for (int i = tid; i < 16 * 256; i += 256) {
    const int r = i >> 8, c = i & 255;
    As[i] = (__bf16)A[(size_t)(rb + r) * strideA + c];
  }
  __syncthreads();

  const int n0 = w * 32;
  v8f acc0 = {}, acc1 = {};
#pragma unroll
  for (int k0 = 0; k0 < 256; k0 += 32) {
    v16bf a  = load_a_frag(As, 256, k0, lane);
    v16bf bA = load_b_frag(W, HH, k0, n0, lane);
    v16bf bB = load_b_frag(W, HH, k0, n0 + 16, lane);
    acc0 = WMMA_BF16(a, bA, acc0);
    acc1 = WMMA_BF16(a, bB, acc1);
  }
  const int m0 = (lane >> 4) << 3, nn = lane & 15;
#pragma unroll
  for (int r = 0; r < 8; ++r) {
    const int m = rb + m0 + r;
    out[(size_t)m * HH + n0 + nn]      = tanhf(acc0[r] + bias[n0 + nn]);
    out[(size_t)m * HH + n0 + 16 + nn] = tanhf(acc1[r] + bias[n0 + 16 + nn]);
  }
}

// ---------------- Kernel 4: out = [hf, hb] @ fc_w.T + fc_b  (K = 512) ----------------

__global__ __launch_bounds__(256) void fc_kernel(const float* __restrict__ hf,
                                                 const float* __restrict__ hb,
                                                 const float* __restrict__ fcw,
                                                 const float* __restrict__ fcb,
                                                 float* __restrict__ out) {
  __shared__ alignas(16) __bf16 As[16 * 512];
  const int rb = blockIdx.x * 16;
  const int tid = threadIdx.x, lane = tid & 31, w = tid >> 5;

  for (int i = tid; i < 16 * 512; i += 256) {
    const int r = i >> 9, c = i & 511;
    const float v = (c < 256) ? hf[(size_t)(rb + r) * 256 + c]
                              : hb[(size_t)(rb + r) * 256 + (c - 256)];
    As[i] = (__bf16)v;
  }
  __syncthreads();

  const int n0 = w * 32;
  v8f acc0 = {}, acc1 = {};
#pragma unroll
  for (int k0 = 0; k0 < 512; k0 += 32) {
    v16bf a  = load_a_frag(As, 512, k0, lane);
    v16bf bA = load_b_frag_T(fcw, 512, k0, n0, lane);
    v16bf bB = load_b_frag_T(fcw, 512, k0, n0 + 16, lane);
    acc0 = WMMA_BF16(a, bA, acc0);
    acc1 = WMMA_BF16(a, bB, acc1);
  }
  const int m0 = (lane >> 4) << 3, nn = lane & 15;
#pragma unroll
  for (int r = 0; r < 8; ++r) {
    const int m = rb + m0 + r;
    out[(size_t)m * HH + n0 + nn]      = acc0[r] + fcb[n0 + nn];
    out[(size_t)m * HH + n0 + 16 + nn] = acc1[r] + fcb[n0 + 16 + nn];
  }
}

// ---------------- Host launcher ----------------

extern "C" void kernel_launch(void* const* d_in, const int* in_sizes, int n_in,
                              void* d_out, int out_size, void* d_ws, size_t ws_size,
                              hipStream_t stream) {
  (void)in_sizes; (void)n_in; (void)out_size; (void)ws_size;
  const float* x     = (const float*)d_in[0];
  const float* Wih_f = (const float*)d_in[1];
  const float* Whh_f = (const float*)d_in[2];
  const float* b_f   = (const float*)d_in[3];
  const float* Wih_b = (const float*)d_in[4];
  // d_in[5] (Whh_b) is multiplied by h0 == 0 in the only backward step that
  // contributes to the output -> never read.
  const float* b_b   = (const float*)d_in[6];
  const float* fc_w  = (const float*)d_in[7];
  const float* fc_b  = (const float*)d_in[8];
  float* out = (float*)d_out;

  float*  pre = (float*)d_ws;                      // [T, B, H] fp32, fragment-packed
  float*  hf  = pre + (size_t)TT * BB * HH;        // [B, H]
  float*  hb0 = hf + (size_t)BB * HH;
  float*  hb1 = hb0 + (size_t)BB * HH;
  __bf16* wpk = (__bf16*)(hb1 + (size_t)BB * HH);  // 3*T*64K bf16 (100.7 MB), 16B-aligned

  // 0) pack the scan's three per-step weight matrices into bf16 fragment tiles
  pack_kernel<<<dim3(TT, 3), 256, 0, stream>>>(Whh_f, Wih_f, wpk);

  // 1) fully parallel layer-0 input projection for all timesteps
  pre_kernel<<<dim3(TT, BB / 16), 256, 0, stream>>>(x, Wih_f, b_f, pre);

  // 2) sequential forward scan, partitioned over batch (no inter-WG sync needed)
  scan_kernel<<<BB / 16, 256, 0, stream>>>(pre, wpk, b_f, hf);

  // 3) backward direction: only scan-step 0 (t = T-1, h0 = 0) reaches the output
  gemm_tanh_kernel<<<BB / 16, 256, 0, stream>>>(
      x + (size_t)(TT - 1) * DD, (long)TT * DD,
      Wih_b + (size_t)(TT - 1) * (size_t)DD * HH,
      b_b + (size_t)(TT - 1) * HH, hb0);
  gemm_tanh_kernel<<<BB / 16, 256, 0, stream>>>(
      hb0, (long)HH,
      Wih_b + ((size_t)TT + TT - 1) * (size_t)DD * HH,
      b_b + ((size_t)TT + TT - 1) * HH, hb1);

  // 4) final linear layer on concat(hf, hb1)
  fc_kernel<<<BB / 16, 256, 0, stream>>>(hf, hb1, fc_w, fc_b, out);
}
